// GATEncoder_87986700026317
// MI455X (gfx1250) — matrix operand order repaired and checked by reading, not compiled, not run on hardware
//
#include <hip/hip_runtime.h>
#include <hip/hip_bf16.h>

typedef __attribute__((ext_vector_type(16))) __bf16 v16bf;
typedef __attribute__((ext_vector_type(8)))  float  v8f;

#define HEADS 4
#define HID   128
#define FDIM  512            // HEADS*HID
#define NEG_SLOPE 0.2f
#define EPSV 1e-16f

// ---- ordered-uint encoding for float atomic max (handles negatives) ----
__device__ __forceinline__ unsigned fenc(float f) {
    unsigned u = __float_as_uint(f);
    return (u & 0x80000000u) ? ~u : (u | 0x80000000u);
}
__device__ __forceinline__ float fdec(unsigned u) {
    return __uint_as_float((u & 0x80000000u) ? (u ^ 0x80000000u) : ~u);
}
#define ENC_NEGINF 0x007FFFFFu   // fenc(-inf)

// ======================================================================
// Pack A [M x K] f32 -> per-lane bf16 WMMA fragments.
// Fragment (m_tile, k_tile): lane l, elem e holds
//   A[m_tile*16 + (l&15)][k_tile*32 + e + (e&8) + (l>=16?8:0)]
// Stored at Asw[((m_tile*ktiles + k_tile)*32 + l)*16 + e]  (32B/lane, coalesced)
// ======================================================================
__global__ void pack_a_bf16(const float* __restrict__ A, __bf16* __restrict__ Asw,
                            int M, int K, int total_tiles) {
    int tid  = blockIdx.x * blockDim.x + threadIdx.x;
    int lane = tid & 31;
    int t    = tid >> 5;
    if (t >= total_tiles) return;
    int ktiles = K >> 5;
    int mt = t / ktiles, kt = t % ktiles;
    int row   = mt * 16 + (lane & 15);
    int kbase = kt * 32 + ((lane >= 16) ? 8 : 0);
    __bf16* outp = Asw + ((size_t)t * 32 + lane) * 16;
    if (row < M) {
        const float* ar = A + (size_t)row * K + kbase;
#pragma unroll
        for (int e = 0; e < 16; ++e) outp[e] = (__bf16)ar[e + (e & 8)];
    } else {
#pragma unroll
        for (int e = 0; e < 16; ++e) outp[e] = (__bf16)0.f;
    }
}

// ======================================================================
// Pack B [K x 512] f32 -> per-lane bf16 WMMA fragments.
// Fragment (k_tile, c_tile of 16 cols): lane l, elem e holds
//   B[k_tile*32 + e + (e&8) + (l>=16?8:0)][c_tile*16 + (l&15)]
// Stored at Bsw[((k_tile*32 + c_tile)*32 + l)*16 + e]
// ======================================================================
__global__ void pack_b_bf16(const float* __restrict__ B, __bf16* __restrict__ Bsw,
                            int K) {
    int tid  = blockIdx.x * blockDim.x + threadIdx.x;
    int lane = tid & 31;
    int t    = tid >> 5;             // t = kt*32 + ct
    int kt = t >> 5, ct = t & 31;
    if (kt >= (K >> 5)) return;
    int col   = ct * 16 + (lane & 15);
    int kbase = kt * 32 + ((lane >= 16) ? 8 : 0);
    __bf16* outp = Bsw + ((size_t)t * 32 + lane) * 16;
    const float* bp = B + (size_t)kbase * FDIM + col;
#pragma unroll
    for (int e = 0; e < 16; ++e) outp[e] = (__bf16)bp[(size_t)(e + (e & 8)) * FDIM];
}

// ======================================================================
// GEMM on pre-packed fragments: C[M x 512] = A * B, f32 accumulate.
// One block = one 16-row tile; wave w (of 8) computes cols [w*64, w*64+64).
// Inner loop: 1 A-frag load + 4 B-frag loads + 4 WMMAs per 32-deep k step.
// ======================================================================
__global__ __launch_bounds__(256) void gemm_bf16_wmma_packed(
    const __bf16* __restrict__ Asw, const __bf16* __restrict__ Bsw,
    float* __restrict__ C, int M, int K) {
    const int wave  = threadIdx.x >> 5;    // col group (64 cols)
    const int lane  = threadIdx.x & 31;
    const int lhalf = (lane >= 16);
    const int lmod  = lane & 15;
    const int row_tile = blockIdx.x;
    const int ktiles = K >> 5;

    const __bf16* abase = Asw + ((size_t)row_tile * ktiles * 32 + lane) * 16;
    const __bf16* bbase = Bsw + ((size_t)wave * 4 * 32 + lane) * 16;

    v8f acc0 = {}, acc1 = {}, acc2 = {}, acc3 = {};
    for (int kt = 0; kt < ktiles; ++kt) {
        v16bf a = *(const v16bf*)(abase + (size_t)kt * 32 * 16);
        const __bf16* bk = bbase + (size_t)kt * 32 * 32 * 16;
        v16bf b0 = *(const v16bf*)(bk);
        v16bf b1 = *(const v16bf*)(bk + 32 * 16);
        v16bf b2 = *(const v16bf*)(bk + 2 * 32 * 16);
        v16bf b3 = *(const v16bf*)(bk + 3 * 32 * 16);
        acc0 = __builtin_amdgcn_wmma_f32_16x16x32_bf16(false, a, false, b0, (short)0, acc0, false, false);
        acc1 = __builtin_amdgcn_wmma_f32_16x16x32_bf16(false, a, false, b1, (short)0, acc1, false, false);
        acc2 = __builtin_amdgcn_wmma_f32_16x16x32_bf16(false, a, false, b2, (short)0, acc2, false, false);
        acc3 = __builtin_amdgcn_wmma_f32_16x16x32_bf16(false, a, false, b3, (short)0, acc3, false, false);
    }

    // C/D layout: VGPR v, lane l -> M = v + (l>=16?8:0), N = l%16
    float* ctile = C + (size_t)(row_tile * 16) * FDIM + wave * 64 + lmod;
#pragma unroll
    for (int v = 0; v < 8; ++v) {
        int r = v + (lhalf ? 8 : 0);
        if (row_tile * 16 + r < M) {
            float* cr = ctile + (size_t)r * FDIM;
            cr[0]  = acc0[v];
            cr[16] = acc1[v];
            cr[32] = acc2[v];
            cr[48] = acc3[v];
        }
    }
}

// ======================================================================
// Per-(node,head) attention score dots: a_s = <h, att_src>, a_d = <h, att_dst>
// ======================================================================
__global__ void attn_dot(const float* __restrict__ h,
                         const float* __restrict__ att_src,
                         const float* __restrict__ att_dst,
                         float* __restrict__ a_s, float* __restrict__ a_d, int NH) {
    int tid = blockIdx.x * blockDim.x + threadIdx.x;
    if (tid >= NH) return;
    int hd = tid & 3;
    int n  = tid >> 2;
    const float* hp = h + (size_t)n * FDIM + hd * HID;
    const float* as = att_src + hd * HID;
    const float* ad = att_dst + hd * HID;
    float s = 0.f, d = 0.f;
#pragma unroll 4
    for (int c = 0; c < HID; ++c) {
        float v = hp[c];
        s += v * as[c];
        d += v * ad[c];
    }
    a_s[tid] = s;
    a_d[tid] = d;
}

// ======================================================================
// Reset per-layer accumulators: accum[N*512]=0, denom[N*4]=0, amax=-inf
// ======================================================================
__global__ void init_layer(float* __restrict__ accum, float* __restrict__ denom,
                           unsigned* __restrict__ amax_u, int N) {
    int tid = blockIdx.x * blockDim.x + threadIdx.x;
    int tot = N * FDIM;
    if (tid < tot) accum[tid] = 0.f;
    if (tid < N * HEADS) { denom[tid] = 0.f; amax_u[tid] = ENC_NEGINF; }
}

__device__ __forceinline__ void edge_endpoints(const long long* __restrict__ ei,
                                               int e, int E, int& src, int& dst) {
    if (e < E) { src = (int)ei[e]; dst = (int)ei[(size_t)E + e]; }
    else       { src = dst = e - E; }      // self loops appended
}

// Pass A: alpha = leaky_relu(a_s[src]+a_d[dst]); segment max over dst.
__global__ void edge_alpha_max(const float* __restrict__ a_s, const float* __restrict__ a_d,
                               const long long* __restrict__ ei, float* __restrict__ ebuf,
                               unsigned* __restrict__ amax_u, int E, int Etot) {
    int tid = blockIdx.x * blockDim.x + threadIdx.x;
    if (tid >= Etot * HEADS) return;
    int e = tid >> 2, hd = tid & 3;
    int src, dst; edge_endpoints(ei, e, E, src, dst);
    float al = a_s[src * HEADS + hd] + a_d[dst * HEADS + hd];
    float lr = (al > 0.f) ? al : NEG_SLOPE * al;
    ebuf[tid] = lr;
    atomicMax(&amax_u[dst * HEADS + hd], fenc(lr));
}

// Pass B: ex = exp(alpha - amax[dst]); segment sum into denom.
__global__ void edge_exp_sum(const unsigned* __restrict__ amax_u,
                             const long long* __restrict__ ei, float* __restrict__ ebuf,
                             float* __restrict__ denom, int E, int Etot) {
    int tid = blockIdx.x * blockDim.x + threadIdx.x;
    if (tid >= Etot * HEADS) return;
    int e = tid >> 2, hd = tid & 3;
    int src, dst; edge_endpoints(ei, e, E, src, dst);
    (void)src;
    float ex = __expf(ebuf[tid] - fdec(amax_u[dst * HEADS + hd]));
    ebuf[tid] = ex;
    atomicAdd(&denom[dst * HEADS + hd], ex);
}

// Pass C: accum[dst] += h[src] * (ex / (denom[dst] + eps)); one block per edge.
__global__ __launch_bounds__(256) void edge_scatter(
    const float* __restrict__ h, const float* __restrict__ ebuf,
    const float* __restrict__ denom, const long long* __restrict__ ei,
    float* __restrict__ accum, int E) {
    int e = blockIdx.x;
    int src, dst; edge_endpoints(ei, e, E, src, dst);
    int elem = threadIdx.x * 2;            // 512 channels / 256 threads
    int hd = elem >> 7;
    float w = ebuf[(size_t)e * HEADS + hd] / (denom[(size_t)dst * HEADS + hd] + EPSV);
    const float2 hv = *(const float2*)(h + (size_t)src * FDIM + elem);
    float* ac = accum + (size_t)dst * FDIM + elem;
    atomicAdd(ac,     hv.x * w);
    atomicAdd(ac + 1, hv.y * w);
}

// Pass D: mean over heads + bias + relu -> [N, HID]
__global__ void head_mean_bias_relu(const float* __restrict__ accum,
                                    const float* __restrict__ bias,
                                    float* __restrict__ out, int tot /* N*HID */) {
    int tid = blockIdx.x * blockDim.x + threadIdx.x;
    if (tid >= tot) return;
    int n = tid >> 7, c = tid & 127;
    const float* ap = accum + (size_t)n * FDIM + c;
    float v = 0.25f * (ap[0] + ap[HID] + ap[2 * HID] + ap[3 * HID]) + bias[c];
    out[tid] = (v > 0.f) ? v : 0.f;
}

// ======================================================================
extern "C" void kernel_launch(void* const* d_in, const int* in_sizes, int n_in,
                              void* d_out, int out_size, void* d_ws, size_t ws_size,
                              hipStream_t stream) {
    const float*     x    = (const float*)d_in[0];
    const long long* ei   = (const long long*)d_in[1];   // int64 edge_index [2,E]
    const float*     W1   = (const float*)d_in[2];
    const float*     as1  = (const float*)d_in[3];
    const float*     ad1  = (const float*)d_in[4];
    const float*     b1   = (const float*)d_in[5];
    const float*     W2   = (const float*)d_in[6];
    const float*     as2  = (const float*)d_in[7];
    const float*     ad2  = (const float*)d_in[8];
    const float*     b2   = (const float*)d_in[9];
    float*           out  = (float*)d_out;

    const int N    = in_sizes[0] / 256;    // F_in = 256
    const int E    = in_sizes[1] / 2;
    const int Etot = E + N;                // with self loops

    const int row_tiles = (N + 15) / 16;
    const int nh        = N * HEADS;
    const int eh        = Etot * HEADS;

    // workspace carve-up (floats unless noted)
    float*    ws     = (float*)d_ws;
    float*    h      = ws;                          // N*512
    float*    accum  = h + (size_t)N * FDIM;        // N*512
    float*    hnext  = accum + (size_t)N * FDIM;    // N*128
    float*    a_s    = hnext + (size_t)N * HID;     // N*4
    float*    a_d    = a_s + (size_t)N * HEADS;     // N*4
    float*    denom  = a_d + (size_t)N * HEADS;     // N*4
    unsigned* amax_u = (unsigned*)(denom + (size_t)N * HEADS);  // N*4
    float*    ebuf   = (float*)(amax_u + (size_t)N * HEADS);    // Etot*4
    __bf16*   asw    = (__bf16*)(ebuf + (size_t)Etot * HEADS);  // row_tiles*8 tiles * 512
    __bf16*   bsw    = asw + (size_t)row_tiles * 8 * 512;       // 8*32 tiles * 512

    // ---------------- Layer 1 (K = 256) ----------------
    {
        const int K = 256, ktiles = K >> 5;
        const int a_tiles = row_tiles * ktiles;
        pack_a_bf16<<<(a_tiles * 32 + 255) / 256, 256, 0, stream>>>(x, asw, N, K, a_tiles);
        pack_b_bf16<<<(ktiles * 32 * 32 + 255) / 256, 256, 0, stream>>>(W1, bsw, K);
        gemm_bf16_wmma_packed<<<row_tiles, 256, 0, stream>>>(asw, bsw, h, N, K);
    }
    attn_dot<<<(nh + 255) / 256, 256, 0, stream>>>(h, as1, ad1, a_s, a_d, nh);
    init_layer<<<((size_t)N * FDIM + 255) / 256, 256, 0, stream>>>(accum, denom, amax_u, N);
    edge_alpha_max<<<(eh + 255) / 256, 256, 0, stream>>>(a_s, a_d, ei, ebuf, amax_u, E, Etot);
    edge_exp_sum<<<(eh + 255) / 256, 256, 0, stream>>>(amax_u, ei, ebuf, denom, E, Etot);
    edge_scatter<<<Etot, 256, 0, stream>>>(h, ebuf, denom, ei, accum, E);
    head_mean_bias_relu<<<((size_t)N * HID + 255) / 256, 256, 0, stream>>>(accum, b1, hnext, N * HID);

    // ---------------- Layer 2 (K = 128) ----------------
    {
        const int K = 128, ktiles = K >> 5;
        const int a_tiles = row_tiles * ktiles;
        pack_a_bf16<<<(a_tiles * 32 + 255) / 256, 256, 0, stream>>>(hnext, asw, N, K, a_tiles);
        pack_b_bf16<<<(ktiles * 32 * 32 + 255) / 256, 256, 0, stream>>>(W2, bsw, K);
        gemm_bf16_wmma_packed<<<row_tiles, 256, 0, stream>>>(asw, bsw, h, N, K);
    }
    attn_dot<<<(nh + 255) / 256, 256, 0, stream>>>(h, as2, ad2, a_s, a_d, nh);
    init_layer<<<((size_t)N * FDIM + 255) / 256, 256, 0, stream>>>(accum, denom, amax_u, N);
    edge_alpha_max<<<(eh + 255) / 256, 256, 0, stream>>>(a_s, a_d, ei, ebuf, amax_u, E, Etot);
    edge_exp_sum<<<(eh + 255) / 256, 256, 0, stream>>>(amax_u, ei, ebuf, denom, E, Etot);
    edge_scatter<<<Etot, 256, 0, stream>>>(h, ebuf, denom, ei, accum, E);
    head_mean_bias_relu<<<((size_t)N * HID + 255) / 256, 256, 0, stream>>>(accum, b2, out, N * HID);

    (void)n_in; (void)out_size; (void)ws_size;
}
